// AttentionHead_11209864642787
// MI455X (gfx1250) — compile-verified, matrix operand order checked by reading, not run on hardware
//
#include <hip/hip_runtime.h>
#include <hip/hip_bf16.h>
#include <math.h>

typedef __attribute__((ext_vector_type(16))) __bf16 v16bf;
typedef __attribute__((ext_vector_type(8)))  __bf16 v8bf;
typedef __attribute__((ext_vector_type(8)))  float  v8f;
typedef __attribute__((ext_vector_type(4)))  float  v4f;

#define NB 8
#define NT 2048
#define NC 1024
#define NH 64

// Workspace layout (bf16 elements):
//  Wt  [192][1024]  packed transposed weights: rows 0-63 Wq^T, 64-127 Wk^T, 128-191 Wv^T
//  Q   [B*T][64]    row-major
//  K   [B*T][64]    row-major
//  Vt  [B][64][T]   V transposed per batch (A-frag of V^T contiguous along t)
static constexpr size_t WT_ELEMS = 192 * 1024;
static constexpr size_t QK_ELEMS = (size_t)NB * NT * NH;
static constexpr size_t Q_OFF  = WT_ELEMS;
static constexpr size_t K_OFF  = Q_OFF + QK_ELEMS;
static constexpr size_t VT_OFF = K_OFF + QK_ELEMS;

// ---------------------------------------------------------------------------
// Kernel 0: pack Wq/Wk/Wv (fp32 [C,64] row-major) -> Wt (bf16 [192][1024], transposed)
// ---------------------------------------------------------------------------
__global__ void wt_pack(const float* __restrict__ Wq, const float* __restrict__ Wk,
                        const float* __restrict__ Wv, __bf16* __restrict__ Wt) {
  int idx = blockIdx.x * blockDim.x + threadIdx.x;   // = n*1024 + c
  if (idx >= 192 * 1024) return;
  int n = idx >> 10;
  int c = idx & 1023;
  const float* src = (n < 64) ? Wq : ((n < 128) ? Wk : Wv);
  Wt[idx] = (__bf16)src[c * 64 + (n & 63)];
}

// ---------------------------------------------------------------------------
// Kernel 1: QKV projection. One 16-row M-tile per block, 3 waves (Q,K,V).
// Each wave: 4 accum tiles (16x64 output), K-loop 1024 in steps of 32.
// ---------------------------------------------------------------------------
__global__ __launch_bounds__(96) void qkv_proj(const float* __restrict__ x,
                                               const __bf16* __restrict__ Wt,
                                               __bf16* __restrict__ Q,
                                               __bf16* __restrict__ K,
                                               __bf16* __restrict__ Vt) {
  const int lane = threadIdx.x & 31;
  const int wave = threadIdx.x >> 5;      // 0 -> Q, 1 -> K, 2 -> V
  const int nl   = lane & 15;
  const int hi   = lane >> 4;
  const int m0   = blockIdx.x * 16;

  v8f acc[4] = {};
  const float* xrow = x + (size_t)(m0 + nl) * NC;

  for (int kk = 0; kk < NC; kk += 32) {
    const int kb = kk + hi * 8;
    // A-fragment: x rows, converted fp32 -> bf16 in registers
    v4f x0 = *(const v4f*)(xrow + kb);
    v4f x1 = *(const v4f*)(xrow + kb + 4);
    v4f x2 = *(const v4f*)(xrow + kb + 16);
    v4f x3 = *(const v4f*)(xrow + kb + 20);
    v16bf a;
#pragma unroll
    for (int i = 0; i < 4; i++) {
      a[i]      = (__bf16)x0[i];
      a[4 + i]  = (__bf16)x1[i];
      a[8 + i]  = (__bf16)x2[i];
      a[12 + i] = (__bf16)x3[i];
    }
#pragma unroll
    for (int nt = 0; nt < 4; nt++) {
      const int n = wave * 64 + nt * 16 + nl;          // packed-weight row
      const __bf16* wrow = Wt + (size_t)n * NC + kb;
      v8bf b0 = *(const v8bf*)(wrow);
      v8bf b1 = *(const v8bf*)(wrow + 16);
      v16bf bf;
#pragma unroll
      for (int i = 0; i < 8; i++) { bf[i] = b0[i]; bf[8 + i] = b1[i]; }
      acc[nt] = __builtin_amdgcn_wmma_f32_16x16x32_bf16(
          false, a, false, bf, (short)0, acc[nt], false, false);
    }
  }

#pragma unroll
  for (int nt = 0; nt < 4; nt++) {
#pragma unroll
    for (int r = 0; r < 8; r++) {
      const int m  = m0 + r + 8 * hi;
      const int nn = nt * 16 + nl;
      const __bf16 v = (__bf16)acc[nt][r];
      if (wave == 0)      Q[(size_t)m * NH + nn] = v;
      else if (wave == 1) K[(size_t)m * NH + nn] = v;
      else {
        const int b = m / NT, t = m % NT;
        Vt[((size_t)b * NH + nn) * NT + t] = v;        // transposed store
      }
    }
  }
}

// ---------------------------------------------------------------------------
// Kernel 2: flash-style causal attention, fully transposed formulation.
//   S^T = K Q^T   (A = K rows over s, B = Q^T cols over t)
//   O^T = V^T P^T (A = V^T rows over h, B = P^T cols over t)
// The C-frag of S^T is, per lane, exactly the B-frag data of P^T for the
// next WMMA -> softmax stays in registers, no LDS, no barriers.
// One wave per 16-row block.
// ---------------------------------------------------------------------------
__global__ __launch_bounds__(32) void attn(const __bf16* __restrict__ Q,
                                           const __bf16* __restrict__ K,
                                           const __bf16* __restrict__ Vt,
                                           float* __restrict__ out) {
  const int lane = threadIdx.x & 31;
  const int nl   = lane & 15;
  const int hi   = lane >> 4;
  const int bi   = blockIdx.x;
  const int b    = bi >> 7;                           // 128 row-blocks per batch
  const int t0   = (bi & 127) * 16;
  const int tg   = t0 + nl;                           // this lane's output row
  const float scale = 0.03125f;                       // 1/sqrt(1024)

  // Q^T B-fragments: col n = t = nl, k = h chunks at hi*8 and hi*8+16
  const __bf16* qrow = Q + ((size_t)b * NT + t0 + nl) * NH;
  v16bf qb[2];
#pragma unroll
  for (int kk = 0; kk < 2; kk++) {
    const __bf16* p = qrow + kk * 32 + hi * 8;
    v8bf q0 = *(const v8bf*)(p);
    v8bf q1 = *(const v8bf*)(p + 16);
#pragma unroll
    for (int i = 0; i < 8; i++) { qb[kk][i] = q0[i]; qb[kk][8 + i] = q1[i]; }
  }

  float rmax = -__builtin_inff();
  float rsum = 0.0f;
  v8f o[4] = {};                                      // O^T: lane=t, regs=h

  const int nsteps = (t0 + 16 + 31) >> 5;             // causal: only s <= t0+15
  for (int ss = 0; ss < nsteps; ss++) {
    const int s0 = ss * 32;

    // ---- S^T = K Q^T for two 16-wide s-tiles ----
    v8f st[2];
#pragma unroll
    for (int j = 0; j < 2; j++) {
      const __bf16* krow = K + ((size_t)b * NT + s0 + j * 16 + nl) * NH; // A row m = s
      v8f c = {};
#pragma unroll
      for (int kk = 0; kk < 2; kk++) {
        const __bf16* p = krow + kk * 32 + hi * 8;
        v8bf k0 = *(const v8bf*)(p);
        v8bf k1 = *(const v8bf*)(p + 16);
        v16bf kf;
#pragma unroll
        for (int i = 0; i < 8; i++) { kf[i] = k0[i]; kf[8 + i] = k1[i]; }
        c = __builtin_amdgcn_wmma_f32_16x16x32_bf16(
            false, kf, false, qb[kk], (short)0, c, false, false);
      }
      st[j] = c;
    }

    // ---- causal mask + scale + in-register row max (row t = nl) ----
    float m = -__builtin_inff();
#pragma unroll
    for (int j = 0; j < 2; j++) {
#pragma unroll
      for (int r = 0; r < 8; r++) {
        const int sg = s0 + j * 16 + r + 8 * hi;      // s index of this slot
        float v = st[j][r] * scale;
        v = (sg <= tg) ? v : -__builtin_inff();
        st[j][r] = v;
        m = fmaxf(m, v);
      }
    }
    m = fmaxf(m, __shfl_xor(m, 16, 32));              // combine the two s-halves

    const float nmax = fmaxf(rmax, m);
    const float corr = __expf(rmax - nmax);           // 0 on first step
    rmax = nmax;

    // ---- P = exp(S - max): stays lane-local, directly forms P^T B-frag ----
    float ps = 0.0f;
    v16bf pb;
#pragma unroll
    for (int r = 0; r < 8; r++) {
      const float p0 = __expf(st[0][r] - nmax);       // exp(-inf)=0 for masked
      const float p1 = __expf(st[1][r] - nmax);
      ps += p0 + p1;
      pb[r]     = (__bf16)p0;                         // k = hi*8 + r
      pb[8 + r] = (__bf16)p1;                         // k = hi*8 + 16 + r
    }
    ps += __shfl_xor(ps, 16, 32);
    rsum = rsum * corr + ps;

    // ---- O^T = corr*O^T + V^T P^T ----
#pragma unroll
    for (int nt = 0; nt < 4; nt++) {
#pragma unroll
      for (int r = 0; r < 8; r++) o[nt][r] *= corr;   // lane-uniform scalar
      const __bf16* vrow = Vt + ((size_t)b * NH + nt * 16 + nl) * NT + s0 + hi * 8;
      v8bf v0 = *(const v8bf*)(vrow);
      v8bf v1 = *(const v8bf*)(vrow + 16);
      v16bf vf;
#pragma unroll
      for (int i = 0; i < 8; i++) { vf[i] = v0[i]; vf[8 + i] = v1[i]; }
      o[nt] = __builtin_amdgcn_wmma_f32_16x16x32_bf16(
          false, vf, false, pb, (short)0, o[nt], false, false);
    }
  }

  // ---- finalize: lane-uniform 1/rsum, contiguous coalesced stores ----
  const float inv = 1.0f / rsum;
  float* orow = out + ((size_t)b * NT + tg) * NH;     // h = nt*16 + 8*hi + r
#pragma unroll
  for (int nt = 0; nt < 4; nt++) {
    v4f lo, hi4;
#pragma unroll
    for (int r = 0; r < 4; r++) { lo[r] = o[nt][r] * inv; hi4[r] = o[nt][4 + r] * inv; }
    *(v4f*)(orow + nt * 16 + 8 * hi)     = lo;
    *(v4f*)(orow + nt * 16 + 8 * hi + 4) = hi4;
  }
}

// ---------------------------------------------------------------------------
extern "C" void kernel_launch(void* const* d_in, const int* in_sizes, int n_in,
                              void* d_out, int out_size, void* d_ws, size_t ws_size,
                              hipStream_t stream) {
  const float* x  = (const float*)d_in[0];
  const float* Wk = (const float*)d_in[1];
  const float* Wq = (const float*)d_in[2];
  const float* Wv = (const float*)d_in[3];

  __bf16* ws = (__bf16*)d_ws;
  __bf16* Wt = ws;            // [192][1024]
  __bf16* Qb = ws + Q_OFF;    // [B*T][64]
  __bf16* Kb = ws + K_OFF;    // [B*T][64]
  __bf16* Vt = ws + VT_OFF;   // [B][64][T]

  wt_pack<<<(192 * 1024) / 256, 256, 0, stream>>>(Wq, Wk, Wv, Wt);
  qkv_proj<<<(NB * NT) / 16, 96, 0, stream>>>(x, Wt, Qb, Kb, Vt);
  attn<<<(NB * NT) / 16, 32, 0, stream>>>(Qb, Kb, Vt, (float*)d_out);
}
